// GDRA_SGC_74869869904021
// MI455X (gfx1250) — matrix operand order repaired and checked by reading, not compiled, not run on hardware
//
#include <hip/hip_runtime.h>
#include <hip/hip_bf16.h>

typedef __attribute__((ext_vector_type(16))) __bf16 v16bf;
typedef __attribute__((ext_vector_type(8)))  __bf16 v8bf;
typedef __attribute__((ext_vector_type(8)))  float  v8f;

#define LDT 40   // padded LDS row stride (bf16 elems) for 32-wide K tiles -> conflict-free b128

// ---------------- WMMA fragment helpers (wave32, 16x16x32 bf16) ----------------

// A (16xK=32): lanes 0-15 row M=lane, K 0..7 & 16..23 ; lanes 16-31 row M=lane-16, K 8..15 & 24..31
__device__ __forceinline__ v16bf load_frag_a(const __bf16* s, int mbase, int lane) {
  const int r = lane & 15, hh = lane >> 4;
  const __bf16* p = s + (mbase + r) * LDT + hh * 8;
  v8bf lo = *(const v8bf*)p;
  v8bf hi = *(const v8bf*)(p + 16);
  return __builtin_shufflevector(lo, hi, 0,1,2,3,4,5,6,7,8,9,10,11,12,13,14,15);
}

// B (K=32 x 16) from Bt stored [N][K]: lanes 0-15 col N=lane K 0..7 & 16..23 ; lanes 16-31 K 8..15 & 24..31
__device__ __forceinline__ v16bf load_frag_b(const __bf16* s, int nbase, int lane) {
  const int r = lane & 15, hh = lane >> 4;
  const __bf16* p = s + (nbase + r) * LDT + hh * 8;
  v8bf lo = *(const v8bf*)p;
  v8bf hi = *(const v8bf*)(p + 16);
  return __builtin_shufflevector(lo, hi, 0,1,2,3,4,5,6,7,8,9,10,11,12,13,14,15);
}

// 128x32 bf16 tile, 256 threads, two 16B chunks per thread
__device__ __forceinline__ void fetch_tile(const __bf16* __restrict__ g, size_t ld,
                                           int row0, int col0, int tid,
                                           uint4& d0, uint4& d1) {
  const int c0 = tid, c1 = tid + 256;
  d0 = *(const uint4*)(g + (size_t)(row0 + (c0 >> 2)) * ld + col0 + (c0 & 3) * 8);
  d1 = *(const uint4*)(g + (size_t)(row0 + (c1 >> 2)) * ld + col0 + (c1 & 3) * 8);
}

__device__ __forceinline__ void store_tile(__bf16* s, int tid, const uint4& d0, const uint4& d1) {
  const int c0 = tid, c1 = tid + 256;
  *(uint4*)(s + (c0 >> 2) * LDT + (c0 & 3) * 8) = d0;
  *(uint4*)(s + (c1 >> 2) * LDT + (c1 & 3) * 8) = d1;
}

__device__ __forceinline__ void store_tile_masked(__bf16* s, int tid, uint4 d0, uint4 d1,
                                                  bool k0, bool k1) {
  if (!k0) d0 = make_uint4(0u, 0u, 0u, 0u);
  if (!k1) d1 = make_uint4(0u, 0u, 0u, 0u);
  store_tile(s, tid, d0, d1);
}

// ---------------- Main bf16 WMMA GEMM: C[MxN] = A[MxK] @ Bt[NxK]^T ----------------
// Single-barrier double-buffered pipeline; optionally writes row-major bf16 C
// and/or column-major bf16 C^T (= the next GEMM's Bt operand).

template<bool RM, bool CM>
__global__ __launch_bounds__(256)
void gemm_bf16(const __bf16* __restrict__ A, const __bf16* __restrict__ Bt,
               __bf16* __restrict__ Crm, __bf16* __restrict__ Ccm,
               int M, int N, int K) {
  __shared__ __bf16 sA[2][128 * LDT];
  __shared__ __bf16 sB[2][128 * LDT];
  const int tid = threadIdx.x;
  const int lane = tid & 31, wave = tid >> 5;
  const int wm = (wave & 1) * 64, wn = (wave >> 1) * 32;
  const int gm0 = blockIdx.y * 128, gn0 = blockIdx.x * 128;

  v8f acc[4][2];
#pragma unroll
  for (int i = 0; i < 4; ++i)
#pragma unroll
    for (int j = 0; j < 2; ++j)
#pragma unroll
      for (int k = 0; k < 8; ++k) acc[i][j][k] = 0.0f;

  const int nK = K >> 5;
  uint4 a0, a1, b0, b1;
  // prologue: tile 0 -> LDS buf0; tile 1 -> regs
  fetch_tile(A,  K, gm0, 0, tid, a0, a1);
  fetch_tile(Bt, K, gn0, 0, tid, b0, b1);
  store_tile(sA[0], tid, a0, a1);
  store_tile(sB[0], tid, b0, b1);
  if (nK > 1) {
    fetch_tile(A,  K, gm0, 32, tid, a0, a1);
    fetch_tile(Bt, K, gn0, 32, tid, b0, b1);
  }
  __syncthreads();

  for (int kt = 0; kt < nK; ++kt) {
    const int cur = kt & 1;
    // regs hold tile kt+1: stage into the other buffer (safe: last reads of it
    // were at iteration kt-1, fenced by the barrier below)
    if (kt + 1 < nK) {
      store_tile(sA[cur ^ 1], tid, a0, a1);
      store_tile(sB[cur ^ 1], tid, b0, b1);
    }
    // issue fetch of tile kt+2 (latency hidden behind this iteration's math)
    if (kt + 2 < nK) {
      fetch_tile(A,  K, gm0, (kt + 2) << 5, tid, a0, a1);
      fetch_tile(Bt, K, gn0, (kt + 2) << 5, tid, b0, b1);
    }
    v16bf bf0 = load_frag_b(sB[cur], wn,      lane);
    v16bf bf1 = load_frag_b(sB[cur], wn + 16, lane);
#pragma unroll
    for (int mt = 0; mt < 4; ++mt) {
      v16bf af = load_frag_a(sA[cur], wm + mt * 16, lane);
      acc[mt][0] = __builtin_amdgcn_wmma_f32_16x16x32_bf16(false, af, false, bf0,
                                                           (short)0, acc[mt][0], false, false);
      acc[mt][1] = __builtin_amdgcn_wmma_f32_16x16x32_bf16(false, af, false, bf1,
                                                           (short)0, acc[mt][1], false, false);
    }
    __syncthreads();
  }

  const int r = lane & 15, hh = lane >> 4;
#pragma unroll
  for (int mt = 0; mt < 4; ++mt)
#pragma unroll
    for (int nt = 0; nt < 2; ++nt) {
      const int mb = gm0 + wm + mt * 16 + hh * 8;   // VGPR v -> row mb+v
      const int n  = gn0 + wn + nt * 16 + r;
      v8f c = acc[mt][nt];
      if (CM) {  // packed 16B store along M
        v8bf o;
#pragma unroll
        for (int v = 0; v < 8; ++v) o[v] = (__bf16)c[v];
        *(v8bf*)(Ccm + (size_t)n * M + mb) = o;
      }
      if (RM) {
#pragma unroll
        for (int v = 0; v < 8; ++v) Crm[(size_t)(mb + v) * N + n] = (__bf16)c[v];
      }
    }
}

// ---------------- Final fused masked SGC GEMM + bias ----------------
// out[4096x256] = (m .* Y1) @ WaT^T + ((1-m) .* Y3) @ WcT^T + bias
// Mask is applied once per A-tile at LDS-staging time (rows zeroed), so the
// WMMA compute loop is mask-free.

__global__ __launch_bounds__(256)
void sgc_final(const __bf16* __restrict__ Y1, const __bf16* __restrict__ Y3,
               const __bf16* __restrict__ WaT, const __bf16* __restrict__ WcT,
               const float* __restrict__ mask, const float* __restrict__ bias,
               float* __restrict__ out) {
  __shared__ __bf16 sA[2][128 * LDT];
  __shared__ __bf16 sB[2][128 * LDT];
  const int tid = threadIdx.x;
  const int lane = tid & 31, wave = tid >> 5;
  const int wm = (wave & 1) * 64, wn = (wave >> 1) * 32;
  const int gm0 = blockIdx.y * 128, gn0 = blockIdx.x * 128;

  // this thread's two staging chunks always map to these two A-tile rows
  const float m0 = mask[gm0 + (tid >> 2)];
  const float m1 = mask[gm0 + ((tid + 256) >> 2)];
  const bool k0p[2] = { m0 != 0.0f, m0 == 0.0f };
  const bool k1p[2] = { m1 != 0.0f, m1 == 0.0f };

  v8f acc[4][2];
#pragma unroll
  for (int i = 0; i < 4; ++i)
#pragma unroll
    for (int j = 0; j < 2; ++j)
#pragma unroll
      for (int k = 0; k < 8; ++k) acc[i][j][k] = 0.0f;

  uint4 a0, a1, b0, b1;
  // prologue: tile 0 (phase 0) -> buf0; tile 1 -> regs
  fetch_tile(Y1,  512, gm0, 0, tid, a0, a1);
  fetch_tile(WaT, 512, gn0, 0, tid, b0, b1);
  store_tile_masked(sA[0], tid, a0, a1, k0p[0], k1p[0]);
  store_tile(sB[0], tid, b0, b1);
  fetch_tile(Y1,  512, gm0, 32, tid, a0, a1);
  fetch_tile(WaT, 512, gn0, 32, tid, b0, b1);
  __syncthreads();

  for (int kt = 0; kt < 32; ++kt) {          // 2 phases x (512/32)
    const int cur = kt & 1;
    if (kt + 1 < 32) {                        // stage tile kt+1 (its phase!)
      const int pn = (kt + 1) >> 4;
      store_tile_masked(sA[cur ^ 1], tid, a0, a1, k0p[pn], k1p[pn]);
      store_tile(sB[cur ^ 1], tid, b0, b1);
    }
    if (kt + 2 < 32) {                        // fetch tile kt+2 from its source
      const int ktn = kt + 2;
      const __bf16* An = (ktn >= 16) ? Y3  : Y1;
      const __bf16* Bn = (ktn >= 16) ? WcT : WaT;
      const int k0n = (ktn & 15) << 5;
      fetch_tile(An, 512, gm0, k0n, tid, a0, a1);
      fetch_tile(Bn, 512, gn0, k0n, tid, b0, b1);
    }
    v16bf bf0 = load_frag_b(sB[cur], wn,      lane);
    v16bf bf1 = load_frag_b(sB[cur], wn + 16, lane);
#pragma unroll
    for (int mt = 0; mt < 4; ++mt) {
      v16bf af = load_frag_a(sA[cur], wm + mt * 16, lane);
      acc[mt][0] = __builtin_amdgcn_wmma_f32_16x16x32_bf16(false, af, false, bf0,
                                                           (short)0, acc[mt][0], false, false);
      acc[mt][1] = __builtin_amdgcn_wmma_f32_16x16x32_bf16(false, af, false, bf1,
                                                           (short)0, acc[mt][1], false, false);
    }
    __syncthreads();
  }

  const int r = lane & 15, hh = lane >> 4;
#pragma unroll
  for (int mt = 0; mt < 4; ++mt)
#pragma unroll
    for (int nt = 0; nt < 2; ++nt) {
      const int mb = gm0 + wm + mt * 16 + hh * 8;
      const int n  = gn0 + wn + nt * 16 + r;
      const float bn = bias[n];
#pragma unroll
      for (int v = 0; v < 8; ++v)
        out[(size_t)(mb + v) * 256 + n] = acc[mt][nt][v] + bn;
    }
}

// ---------------- prep / GAT kernels (small) ----------------

__global__ __launch_bounds__(256)
void cvt_bf16x8(const float* __restrict__ in, __bf16* __restrict__ out, int n) {
  const int i = (blockIdx.x * 256 + threadIdx.x) * 8;
  if (i + 8 <= n) {
    float4 f0 = *(const float4*)(in + i);
    float4 f1 = *(const float4*)(in + i + 4);
    v8bf o;
    o[0] = (__bf16)f0.x; o[1] = (__bf16)f0.y; o[2] = (__bf16)f0.z; o[3] = (__bf16)f0.w;
    o[4] = (__bf16)f1.x; o[5] = (__bf16)f1.y; o[6] = (__bf16)f1.z; o[7] = (__bf16)f1.w;
    *(v8bf*)(out + i) = o;
  }
}

// out[k*rows + i] = (bf16) in[i*cols + k]
__global__ __launch_bounds__(256)
void transpose_cvt(const float* __restrict__ in, __bf16* __restrict__ out, int rows, int cols) {
  const int idx = blockIdx.x * 256 + threadIdx.x;
  if (idx < rows * cols) {
    const int i = idx / cols, k = idx % cols;
    out[(size_t)k * rows + i] = (__bf16)in[idx];
  }
}

// h = x @ W1 (4096x8), dst[i] = h[i,:] . a1[8:16]
__global__ __launch_bounds__(256)
void gat1_h(const float* __restrict__ x, const float* __restrict__ W1,
            const float* __restrict__ a1, float* __restrict__ h, float* __restrict__ dst) {
  const int tid = threadIdx.x;
  const int r = tid >> 3, j = tid & 7;
  const int row = blockIdx.x * 32 + r;
  const float* xr = x + (size_t)row * 512;
  float acc = 0.0f;
  for (int k = 0; k < 512; ++k) acc += xr[k] * W1[k * 8 + j];
  h[row * 8 + j] = acc;
  __shared__ float sh[256];
  sh[tid] = acc * a1[8 + j];
  __syncthreads();
  if (j == 0) {
    float s = 0.0f;
#pragma unroll
    for (int t = 0; t < 8; ++t) s += sh[r * 8 + t];
    dst[row] = s;
  }
}

// single block: stats[0]=max(v), stats[1]=1/sum(exp(v-max))
__global__ __launch_bounds__(256)
void softmax_stats(const float* __restrict__ v, int n, float* __restrict__ stats) {
  __shared__ float sm[256];
  const int tid = threadIdx.x;
  float mx = -1e30f;
  for (int i = tid; i < n; i += 256) mx = fmaxf(mx, v[i]);
  sm[tid] = mx; __syncthreads();
  for (int s = 128; s > 0; s >>= 1) { if (tid < s) sm[tid] = fmaxf(sm[tid], sm[tid + s]); __syncthreads(); }
  const float m = sm[0]; __syncthreads();
  float sum = 0.0f;
  for (int i = tid; i < n; i += 256) sum += expf(v[i] - m);
  sm[tid] = sum; __syncthreads();
  for (int s = 128; s > 0; s >>= 1) { if (tid < s) sm[tid] += sm[tid + s]; __syncthreads(); }
  if (tid == 0) { stats[0] = m; stats[1] = 1.0f / sm[0]; }
}

// h1[i,:] = elu( sum_j adj[i,j] * p_j * h[j,:] ) ; h2[i] = h1[i,:].W2 ; dst2[i] = a2[1]*h2[i]
__global__ __launch_bounds__(256)
void gat1_aggregate(const float* __restrict__ adj, const float* __restrict__ h,
                    const float* __restrict__ dst, const float* __restrict__ stats,
                    const float* __restrict__ W2, const float* __restrict__ a2,
                    float* __restrict__ h1, float* __restrict__ h2, float* __restrict__ dst2) {
  const int i = blockIdx.x, tid = threadIdx.x;
  const float mx = stats[0], inv = stats[1];
  const float* arow = adj + (size_t)i * 4096;
  float acc[8] = {0, 0, 0, 0, 0, 0, 0, 0};
  for (int j = tid; j < 4096; j += 256) {
    const float a = arow[j];
    if (a != 0.0f) {
      const float p = expf(dst[j] - mx) * inv * a;
      const float* hj = h + j * 8;
#pragma unroll
      for (int c = 0; c < 8; ++c) acc[c] += p * hj[c];
    }
  }
  __shared__ float sh[8 * 256];
#pragma unroll
  for (int c = 0; c < 8; ++c) sh[c * 256 + tid] = acc[c];
  __syncthreads();
  for (int s = 128; s > 0; s >>= 1) {
    if (tid < s) {
#pragma unroll
      for (int c = 0; c < 8; ++c) sh[c * 256 + tid] += sh[c * 256 + tid + s];
    }
    __syncthreads();
  }
  if (tid < 8) {
    float v = sh[tid * 256];
    v = (v > 0.0f) ? v : (expf(v) - 1.0f);   // elu
    sh[tid * 256] = v;
    h1[i * 8 + tid] = v;
  }
  __syncthreads();
  if (tid == 0) {
    float s2 = 0.0f;
#pragma unroll
    for (int c = 0; c < 8; ++c) s2 += sh[c * 256] * W2[c];
    h2[i] = s2;
    dst2[i] = a2[1] * s2;
  }
}

// node_score[i] = elu( sum_j adj[i,j] * p2_j * h2[j] ); mask[i] = score > 0.7
__global__ __launch_bounds__(256)
void gat2_aggregate(const float* __restrict__ adj, const float* __restrict__ h2,
                    const float* __restrict__ dst2, const float* __restrict__ stats,
                    float* __restrict__ maskout) {
  const int i = blockIdx.x, tid = threadIdx.x;
  const float mx = stats[0], inv = stats[1];
  const float* arow = adj + (size_t)i * 4096;
  float acc = 0.0f;
  for (int j = tid; j < 4096; j += 256) {
    const float a = arow[j];
    if (a != 0.0f) acc += a * expf(dst2[j] - mx) * inv * h2[j];
  }
  __shared__ float sh[256];
  sh[tid] = acc; __syncthreads();
  for (int s = 128; s > 0; s >>= 1) { if (tid < s) sh[tid] += sh[tid + s]; __syncthreads(); }
  if (tid == 0) {
    float v = sh[0];
    v = (v > 0.0f) ? v : (expf(v) - 1.0f);
    maskout[i] = (v > 0.7f) ? 1.0f : 0.0f;
  }
}

// ---------------- launch ----------------

extern "C" void kernel_launch(void* const* d_in, const int* in_sizes, int n_in,
                              void* d_out, int out_size, void* d_ws, size_t ws_size,
                              hipStream_t stream) {
  const float* x     = (const float*)d_in[0];   // [4096,512]
  const float* adj   = (const float*)d_in[1];   // [4096,4096]
  const float* W1    = (const float*)d_in[2];   // [512,8]
  const float* a1    = (const float*)d_in[3];   // [16]
  const float* W2    = (const float*)d_in[4];   // [8]
  const float* a2    = (const float*)d_in[5];   // [2]
  const float* W_sgc = (const float*)d_in[6];   // [1536,256]
  const float* b_sgc = (const float*)d_in[7];   // [256]
  float* out = (float*)d_out;                   // [4096,256]

  char* ws = (char*)d_ws;
  size_t off = 0;
  auto alloc = [&](size_t bytes) -> char* {
    char* p = ws + off;
    off += (bytes + 255) & ~(size_t)255;
    return p;
  };
  __bf16* adjb = (__bf16*)alloc((size_t)4096 * 4096 * 2);
  __bf16* xT   = (__bf16*)alloc((size_t)512 * 4096 * 2);
  __bf16* y1T  = (__bf16*)alloc((size_t)512 * 4096 * 2);
  __bf16* y2T  = (__bf16*)alloc((size_t)512 * 4096 * 2);
  __bf16* y1b  = (__bf16*)alloc((size_t)4096 * 512 * 2);
  __bf16* y3b  = (__bf16*)alloc((size_t)4096 * 512 * 2);
  __bf16* WaT  = (__bf16*)alloc((size_t)256 * 512 * 2);
  __bf16* WcT  = (__bf16*)alloc((size_t)256 * 512 * 2);
  float* hbuf  = (float*)alloc((size_t)4096 * 8 * 4);
  float* dst1  = (float*)alloc(4096 * 4);
  float* h1    = (float*)alloc((size_t)4096 * 8 * 4);
  float* h2    = (float*)alloc(4096 * 4);
  float* dst2  = (float*)alloc(4096 * 4);
  float* maskb = (float*)alloc(4096 * 4);
  float* st1   = (float*)alloc(256);
  float* st2   = (float*)alloc(256);
  if (off > ws_size) return;  // workspace too small: bail deterministically

  // --- prep: bf16 conversions / transposes ---
  cvt_bf16x8<<<(4096 * 4096) / 2048, 256, 0, stream>>>(adj, adjb, 4096 * 4096);
  transpose_cvt<<<(4096 * 512) / 256, 256, 0, stream>>>(x, xT, 4096, 512);
  transpose_cvt<<<(512 * 256) / 256, 256, 0, stream>>>(W_sgc, WaT, 512, 256);
  transpose_cvt<<<(512 * 256) / 256, 256, 0, stream>>>(W_sgc + (size_t)1024 * 256, WcT, 512, 256);

  // --- GAT scoring path (softmax over dst only; src term cancels) ---
  gat1_h<<<128, 256, 0, stream>>>(x, W1, a1, hbuf, dst1);
  softmax_stats<<<1, 256, 0, stream>>>(dst1, 4096, st1);
  gat1_aggregate<<<4096, 256, 0, stream>>>(adj, hbuf, dst1, st1, W2, a2, h1, h2, dst2);
  softmax_stats<<<1, 256, 0, stream>>>(dst2, 4096, st2);
  gat2_aggregate<<<4096, 256, 0, stream>>>(adj, h2, dst2, st2, maskb);

  // --- chained hop GEMMs: y1 = adj@x, y2 = adj@y1, y3 = adj@y2 (bf16 WMMA) ---
  dim3 g(4, 32);
  gemm_bf16<true,  true ><<<g, 256, 0, stream>>>(adjb, xT,  y1b, y1T, 4096, 512, 4096);
  gemm_bf16<false, true ><<<g, 256, 0, stream>>>(adjb, y1T, nullptr, y2T, 4096, 512, 4096);
  gemm_bf16<true,  false><<<g, 256, 0, stream>>>(adjb, y2T, y3b, nullptr, 4096, 512, 4096);

  // --- fused masked output GEMM + bias (middle hop is identically zero) ---
  sgc_final<<<dim3(2, 32), 256, 0, stream>>>(y1b, y3b, WaT, WcT, maskb, b_sgc, out);
  (void)in_sizes; (void)n_in; (void)out_size;
}
